// TextAwareVisualExtraction_29798483099827
// MI455X (gfx1250) — matrix operand order repaired and checked by reading, not compiled, not run on hardware
//
#include <hip/hip_runtime.h>
#include <hip/hip_bf16.h>
#include <cstdint>

#define BB 32
#define II 256
#define KK 4096
#define DD 1024

typedef __bf16 bf16_t;
typedef bf16_t v16bf __attribute__((ext_vector_type(16)));
typedef float  v8f   __attribute__((ext_vector_type(8)));

// v16bf + plain uint array only (HIP's uint4 has non-trivial ctors -> not union-safe)
union AB16 { v16bf v; unsigned int u[8]; };

// Pack two f32 into packed bf16x2 (truncation) with a single v_perm_b32:
// result bytes = [a.b2, a.b3, b.b2, b.b3]  (a in low half, b in high half).
__device__ __forceinline__ unsigned int pkbf2(float a, float b) {
  return __builtin_amdgcn_perm(__float_as_uint(b), __float_as_uint(a), 0x07060302u);
}
__device__ __forceinline__ unsigned short f2bf(float a) {
  return (unsigned short)(__float_as_uint(a) >> 16);   // truncating f32->bf16
}
__device__ __forceinline__ float bf2f(unsigned int lo16) {
  return __uint_as_float(lo16 << 16);
}

// Byte offset into packed A-fragment LDS for matrix row m, reduction index r (0..31)
// within chunk `chunk`. Contiguous 16-bit A 16x32 layout: lane = m + half*16,
// element e in the lane's v16bf covers K = e + half*16.
__device__ __forceinline__ int apack_off(int m, int r, int chunk) {
  int half = r >> 4;
  int e = r & 15;
  return chunk * 1024 + (m + half * 16) * 32 + e * 2;
}

// ---------------------------------------------------------------------------
// Kernel 1: sim = text @ image^T (scaled, masked), softmax statistics,
// writes unnormalized exp(sim - rowmax) as bf16 to ws, 1/rowsum to ws.
// Grid: (I/16, B), block 256 (8 waves). Wave w owns k-columns [w*512, w*512+512).
// ---------------------------------------------------------------------------
__global__ __launch_bounds__(256)
void k_sim_softmax(const float* __restrict__ txt, const float* __restrict__ img,
                   const unsigned char* __restrict__ mask,
                   const float* __restrict__ temp,
                   unsigned short* __restrict__ attn, float* __restrict__ invs) {
  __shared__ unsigned char sA[32 * 1024] __attribute__((aligned(16)));
  __shared__ float sRed[8 * 16];
  __shared__ float sRowMax[16];

  const int iBase = blockIdx.x * 16;
  const int b     = blockIdx.y;
  const int tid   = threadIdx.x;
  const int w     = tid >> 5;
  const int lane  = tid & 31;
  const int half  = lane >> 4;
  const int ln    = lane & 15;

  // Phase 0: pack 16 x D text tile into A-fragment layout (f32 -> bf16).
  const float* tp = txt + (size_t)(b * II + iBase) * DD;
  for (int idx = tid; idx < 16 * DD; idx += 256) {
    int m = idx >> 10;
    int d = idx & (DD - 1);
    float x = tp[(size_t)m * DD + d];
    *(unsigned short*)(sA + apack_off(m, d & 31, d >> 5)) = f2bf(x);
  }
  __syncthreads();

  const float invT   = 1.0f / fminf(fmaxf(temp[0], 0.0f), 100.0f);
  const float NEGINF = -__builtin_inff();

  float rmax[8];
#pragma unroll
  for (int v = 0; v < 8; ++v) rmax[v] = NEGINF;

  // Phase 1: GEMM over this wave's 512 columns, 4 tiles in flight (A reuse x4).
  for (int g = 0; g < 8; ++g) {
    const int kbase = w * 512 + g * 64;
    v8f acc[4];
#pragma unroll
    for (int t = 0; t < 4; ++t)
#pragma unroll
      for (int v = 0; v < 8; ++v) acc[t][v] = 0.0f;

    const float* rb[4];
#pragma unroll
    for (int t = 0; t < 4; ++t)
      rb[t] = img + (size_t)(b * KK + kbase + t * 16 + ln) * DD;

    for (int c = 0; c < 32; ++c) {
      AB16 a;
      *(uint4*)&a.u[0] = *(const uint4*)(sA + c * 1024 + lane * 32);
      *(uint4*)&a.u[4] = *(const uint4*)(sA + c * 1024 + lane * 32 + 16);
#pragma unroll
      for (int t = 0; t < 4; ++t) {
        const float* gp = rb[t] + c * 32 + half * 16;  // 16 contiguous f32
        float4 f0 = ((const float4*)gp)[0];
        float4 f1 = ((const float4*)gp)[1];
        float4 f2 = ((const float4*)gp)[2];
        float4 f3 = ((const float4*)gp)[3];
        AB16 bm;
        bm.u[0] = pkbf2(f0.x, f0.y); bm.u[1] = pkbf2(f0.z, f0.w);
        bm.u[2] = pkbf2(f1.x, f1.y); bm.u[3] = pkbf2(f1.z, f1.w);
        bm.u[4] = pkbf2(f2.x, f2.y); bm.u[5] = pkbf2(f2.z, f2.w);
        bm.u[6] = pkbf2(f3.x, f3.y); bm.u[7] = pkbf2(f3.z, f3.w);
        acc[t] = __builtin_amdgcn_wmma_f32_16x16x32_bf16(
            false, a.v, false, bm.v, (short)0, acc[t], false, false);
        if (c + 1 < 32) __builtin_prefetch(gp + 32, 0, 0);  // global_prefetch_b8
      }
    }

    // Epilogue: scale, mask, track row max, store scaled sim (bf16) to ws.
#pragma unroll
    for (int t = 0; t < 4; ++t) {
      const int col  = kbase + t * 16 + ln;
      const bool mk  = mask[(size_t)b * KK + col] != 0;
      const size_t rowb = (size_t)(b * II + iBase + half * 8) * KK + col;
#pragma unroll
      for (int v = 0; v < 8; ++v) {
        float s = mk ? acc[t][v] * invT : NEGINF;
        rmax[v] = fmaxf(rmax[v], s);
        attn[rowb + (size_t)v * KK] = f2bf(s);
      }
    }
  }

  // Row-max reduce across the 16 lanes of each half, then across waves.
#pragma unroll
  for (int v = 0; v < 8; ++v) {
    float m = rmax[v];
    for (int o = 8; o > 0; o >>= 1) m = fmaxf(m, __shfl_xor(m, o, 32));
    if (ln == 0) sRed[w * 16 + half * 8 + v] = m;
  }
  __syncthreads();
  if (tid < 16) {
    float m = NEGINF;
    for (int ww = 0; ww < 8; ++ww) m = fmaxf(m, sRed[ww * 16 + tid]);
    sRowMax[tid] = m;
  }
  __syncthreads();

  // Phase 2: exp(s - rowmax) over this wave's 16x512 slice (L2-hot), row sums.
  for (int r = 0; r < 16; ++r) {
    float rm = sRowMax[r];
    if (rm == NEGINF) rm = 0.0f;  // fully-masked row -> exp(-inf)=0
    const size_t off = (size_t)(b * II + iBase + r) * KK + w * 512 + lane * 16;
    uint4* p = (uint4*)(attn + off);
    unsigned int qu[8];
    *(uint4*)&qu[0] = p[0];
    *(uint4*)&qu[4] = p[1];
    float part = 0.0f;
#pragma unroll
    for (int j = 0; j < 8; ++j) {
      float s0 = bf2f(qu[j] & 0xFFFFu);
      float s1 = bf2f(qu[j] >> 16);
      float e0 = __expf(s0 - rm);
      float e1 = __expf(s1 - rm);
      part += e0 + e1;
      qu[j] = pkbf2(e0, e1);
    }
    p[0] = *(const uint4*)&qu[0];
    p[1] = *(const uint4*)&qu[4];
    for (int o = 16; o > 0; o >>= 1) part += __shfl_xor(part, o, 32);
    if (lane == 0) sRed[w * 16 + r] = part;
  }
  __syncthreads();
  if (tid < 16) {
    float tot = 0.0f;
    for (int ww = 0; ww < 8; ++ww) tot += sRed[ww * 16 + tid];
    invs[b * II + iBase + tid] = (tot > 0.0f) ? (1.0f / tot) : 0.0f;
  }
}

// ---------------------------------------------------------------------------
// Kernel 2: out = (expAttn @ image) * invRowSum.
// Grid: (D/128, I/16, B), block 256. Wave w owns a 16x16 output tile
// (cols dsup*128 + w*16 .. +16) reduced over all K=4096.
// ---------------------------------------------------------------------------
__global__ __launch_bounds__(256)
void k_attn_v(const unsigned short* __restrict__ attn, const float* __restrict__ img,
              const float* __restrict__ invs, float* __restrict__ out) {
  __shared__ unsigned char sA[16 * 1024] __attribute__((aligned(16)));
  __shared__ float sInv[16];

  const int dsup  = blockIdx.x;
  const int iBase = blockIdx.y * 16;
  const int b     = blockIdx.z;
  const int tid   = threadIdx.x;
  const int w     = tid >> 5;
  const int lane  = tid & 31;
  const int half  = lane >> 4;
  const int ln    = lane & 15;
  const int dcol  = dsup * 128 + w * 16 + ln;

  if (tid < 16) sInv[tid] = invs[b * II + iBase + tid];

  v8f acc;
#pragma unroll
  for (int v = 0; v < 8; ++v) acc[v] = 0.0f;

  for (int s = 0; s < 8; ++s) {
    __syncthreads();
    // Stage 16 x 512 bf16 attention slab into packed A-fragment layout.
    // K-pairs (even r) map to consecutive e, so move 2 bf16 per uint.
    for (int idx = tid; idx < 4096; idx += 256) {
      int m = idx >> 8;
      int r = (idx & 255) * 2;
      unsigned int val = *(const unsigned int*)(
          attn + (size_t)(b * II + iBase + m) * KK + s * 512 + r);
      int rr = r & 31;
      int e  = rr & 15;
      int hh = rr >> 4;
      *(unsigned int*)(sA + (r >> 5) * 1024 + (m + hh * 16) * 32 + e * 2) = val;
    }
    __syncthreads();

    const int k0 = s * 512;
    for (int c = 0; c < 16; ++c) {
      AB16 a;
      *(uint4*)&a.u[0] = *(const uint4*)(sA + c * 1024 + lane * 32);
      *(uint4*)&a.u[4] = *(const uint4*)(sA + c * 1024 + lane * 32 + 16);
      // B[kappa, n] = image[k0 + c*32 + half*16 + e, dcol]; lanes give 64B rows.
      const float* gp = img + (size_t)(b * KK + k0 + c * 32 + half * 16) * DD + dcol;
      AB16 bm;
#pragma unroll
      for (int j = 0; j < 8; ++j) {
        float f0 = gp[(size_t)(2 * j)     * DD];
        float f1 = gp[(size_t)(2 * j + 1) * DD];
        bm.u[j] = pkbf2(f0, f1);
      }
      __builtin_prefetch(gp + 32 * DD, 0, 0);  // next chunk's rows
      acc = __builtin_amdgcn_wmma_f32_16x16x32_bf16(
          false, a.v, false, bm.v, (short)0, acc, false, false);
    }
  }

#pragma unroll
  for (int v = 0; v < 8; ++v) {
    const int row = iBase + half * 8 + v;
    out[(size_t)(b * II + row) * DD + dcol] = acc[v] * sInv[half * 8 + v];
  }
}

extern "C" void kernel_launch(void* const* d_in, const int* in_sizes, int n_in,
                              void* d_out, int out_size, void* d_ws, size_t ws_size,
                              hipStream_t stream) {
  (void)in_sizes; (void)n_in; (void)out_size; (void)ws_size;
  const float*         img  = (const float*)d_in[0];          // (B,K,D) f32
  const float*         txt  = (const float*)d_in[1];          // (B,I,D) f32
  const unsigned char* mask = (const unsigned char*)d_in[2];  // (B,K) bool8
  const float*         temp = (const float*)d_in[3];          // scalar f32

  unsigned short* attn = (unsigned short*)d_ws;                       // B*I*K bf16 = 64 MB
  float* invs = (float*)((char*)d_ws + (size_t)BB * II * KK * 2);     // B*I f32
  float* out  = (float*)d_out;                                        // (B,I,D) f32

  dim3 g1(II / 16, BB);
  k_sim_softmax<<<g1, dim3(256), 0, stream>>>(txt, img, mask, temp, attn, invs);

  dim3 g2(DD / 128, II / 16, BB);
  k_attn_v<<<g2, dim3(256), 0, stream>>>(attn, img, invs, out);
}
